// LPN_cond_24541443129517
// MI455X (gfx1250) — compile-verified, matrix operand order checked by reading, not run on hardware
//
#include <hip/hip_runtime.h>
#include <hip/hip_bf16.h>

// ---------------------------------------------------------------------------
// LPN_cond forward + exact input-gradient for MI455X (gfx1250).
// Convs (fwd + transposed bwd) as implicit GEMM via V_WMMA_F32_16X16X32_F16.
// Block = 8 waves, output tile 128(M) x 32(N); B-tile staged in LDS as f32
// (async global->LDS fast path when contiguous), A = f16-packed weights from
// L2. Two WMMA per wave per 32-k step (A fragment reused across both n-tiles).
// ---------------------------------------------------------------------------

typedef __attribute__((ext_vector_type(16))) _Float16 v16h;
typedef __attribute__((ext_vector_type(8)))  float    v8f;

#if __has_builtin(__builtin_amdgcn_global_load_async_to_lds_b32)
#define HAS_ASYNC_LDS 1
typedef __attribute__((address_space(1))) int gas_int;   // global (AS1)
typedef __attribute__((address_space(3))) int las_int;   // LDS (AS3)
#endif

enum { F_BIAS = 1, F_SOFTPLUS = 2, F_RELU = 4, F_ACC = 16, F_TRANS = 32 };

// K index held by (lane, half h) of a 16-bit 16x32 fragment (ISA 7.12.2).
__device__ __forceinline__ int frag_k(int lane, int h) {
  int v = h >> 1, p = h & 1;
  return ((lane & 16) ? 8 : 0) + ((v & 4) ? 16 : 0) + ((v & 3) << 1) + p;
}

template <int KW, bool TRANS>
__global__ __launch_bounds__(256) void conv1d_wmma(
    const float* __restrict__ in, const _Float16* __restrict__ wgt,
    const float* __restrict__ bias, float* __restrict__ out,
    int Cin, int Lin, int Cout, int Lout, int stride, int pad, int flags) {
  __shared__ float sBf[32][33];  // 32 k x 32 n, f32 (async copies raw bytes)
  const int b     = blockIdx.y;
  const int nBase = blockIdx.x * 32;
  const int tid   = threadIdx.x;
  const int lane  = tid & 31;    // wave32
  const int wv    = tid >> 5;
  const int mBase = wv * 16;
  const int KK    = Cin * KW;
  const int mA    = mBase + (lane & 15);
  const bool aOk  = (mA < Cout);
  const size_t inB = (size_t)b * Cin * (size_t)Lin;
  const _Float16* wrow = wgt + (size_t)mA * KK;
  const bool fullN = (nBase + 32 <= Lout);

  v8f acc0 = {}, acc1 = {};
  for (int k0 = 0; k0 < KK; k0 += 32) {
    // ---- stage B tile: wave wv owns rows kl = 4*wv .. 4*wv+3 (kl uniform
    // per wave -> ci,kt resolve to scalar ops; KW is a template constant). ----
#pragma unroll
    for (int r = 0; r < 4; ++r) {
      const int kl = wv * 4 + r;
      const int kg = k0 + kl;
      const int ci = kg / KW, kt = kg - ci * KW;  // constant-divisor
      const int n  = lane;
      bool staged = false;
#ifdef HAS_ASYNC_LDS
      // contiguous fast path: stride==1 and full row in bounds ->
      // one global_load_async_to_lds_b32 per lane (ASYNCcnt-tracked DMA).
      if (stride == 1 && kg < KK && fullN) {
        const int l0 = TRANS ? (nBase + pad - kt) : (nBase - pad + kt);
        if (l0 >= 0 && l0 + 32 <= Lin) {
          const float* src = in + inB + (size_t)ci * Lin + l0 + n;
          __builtin_amdgcn_global_load_async_to_lds_b32(
              (gas_int*)(unsigned long long)src,
              (las_int*)(unsigned)(unsigned long long)&sBf[kl][n], 0, 0);
          staged = true;
        }
      }
#endif
      if (!staged) {
        float v = 0.f;
        const int nPos = nBase + n;
        if (kg < KK && nPos < Lout) {
          if (!TRANS) {
            int l = nPos * stride - pad + kt;
            if (l >= 0 && l < Lin) v = in[inB + (size_t)ci * Lin + l];
          } else {  // conv-transpose gather: lo = (li + pad - kt) / stride
            int t = nPos + pad - kt;
            if (t >= 0 && (t % stride) == 0) {
              int lo = t / stride;
              if (lo < Lin) v = in[inB + (size_t)ci * Lin + lo];
            }
          }
        }
        sBf[kl][n] = v;
      }
    }
#ifdef HAS_ASYNC_LDS
    asm volatile("s_wait_asynccnt 0" ::: "memory");
#endif
    __syncthreads();

    if (k0 + 32 < KK) __builtin_prefetch(wrow + k0 + 32, 0, 3);  // global_prefetch_b8

    // ---- fragments: A from packed f16 weights, B0/B1 from LDS (cvt f32->f16) ----
    v16h a, b0, b1;
#pragma unroll
    for (int h = 0; h < 16; ++h) {
      const int kl = frag_k(lane, h);
      const int kg = k0 + kl;
      a[h]  = (aOk && kg < KK) ? wrow[kg] : (_Float16)0.f;
      b0[h] = (_Float16)sBf[kl][lane & 15];
      b1[h] = (_Float16)sBf[kl][16 + (lane & 15)];
    }
    acc0 = __builtin_amdgcn_wmma_f32_16x16x32_f16(false, a, false, b0,
                                                  (short)0, acc0, false, false);
    acc1 = __builtin_amdgcn_wmma_f32_16x16x32_f16(false, a, false, b1,
                                                  (short)0, acc1, false, false);
    __syncthreads();
  }

  // ---- epilogue: bias / softplus / relu, optional accumulate; two n-tiles ----
  const size_t outB = (size_t)b * Cout * (size_t)Lout;
#pragma unroll
  for (int t = 0; t < 2; ++t) {
    const int n = nBase + 16 * t + (lane & 15);
    if (n >= Lout) continue;
    const v8f& av = t ? acc1 : acc0;
#pragma unroll
    for (int r = 0; r < 8; ++r) {
      const int m = mBase + r + ((lane & 16) ? 8 : 0);  // C/D: VGPR r -> M=r / r+8
      if (m < Cout) {
        float v = av[r];
        if (flags & F_BIAS)     v += bias[m];
        if (flags & F_SOFTPLUS) v = (v > 20.f) ? v : log1pf(__expf(v));
        if (flags & F_RELU)     v = fmaxf(v, 0.f);
        size_t o = outB + (size_t)m * Lout + n;
        if (flags & F_ACC) out[o] += v; else out[o] = v;
      }
    }
  }
}

// f32 -> f16 weight pack; transposed pack for conv-transpose (wT[ci][co][kt]).
__global__ void pack_w_kernel(const float* __restrict__ w, _Float16* __restrict__ o,
                              int Co, int Ci, int Kw, int tr) {
  size_t n = (size_t)Co * Ci * Kw;
  for (size_t i = blockIdx.x * 256ull + threadIdx.x; i < n; i += (size_t)gridDim.x * 256ull) {
    if (!tr) { o[i] = (_Float16)w[i]; }
    else {
      int kt = (int)(i % Kw); size_t t = i / Kw;
      int co = (int)(t % Co); int ci = (int)(t / Co);
      o[i] = (_Float16)w[((size_t)co * Ci + ci) * Kw + kt];
    }
  }
}

// ---- small elementwise kernels (grid-stride) ----
__global__ void ew_tilde0(float* u, const float* sg, const float* w, const float* bb, size_t N) {
  size_t total = 512ull * N;
  for (size_t i = blockIdx.x * 256ull + threadIdx.x; i < total; i += (size_t)gridDim.x * 256ull) {
    size_t b = i / N, j = i % N;
    u[i] = sg[b] * w[j] + bb[j];
  }
}
__global__ void ew_mul(float* o, const float* a, const float* b, size_t n) {
  for (size_t i = blockIdx.x * 256ull + threadIdx.x; i < n; i += (size_t)gridDim.x * 256ull)
    o[i] = a[i] * b[i];
}
__global__ void ew_mul_acc(float* o, const float* a, const float* b, size_t n) {
  for (size_t i = blockIdx.x * 256ull + threadIdx.x; i < n; i += (size_t)gridDim.x * 256ull)
    o[i] += a[i] * b[i];
}
__global__ void ew_softplus(float* o, const float* z, size_t n) {
  for (size_t i = blockIdx.x * 256ull + threadIdx.x; i < n; i += (size_t)gridDim.x * 256ull) {
    float v = z[i];
    o[i] = (v > 20.f) ? v : log1pf(__expf(v));
  }
}
__global__ void ew_sigmul(float* o, const float* g, const float* z, size_t n) {
  for (size_t i = blockIdx.x * 256ull + threadIdx.x; i < n; i += (size_t)gridDim.x * 256ull)
    o[i] = g[i] / (1.f + __expf(-z[i]));   // g * sigmoid(z_pre)  (softplus')
}
__global__ void ew_scale(float* o, const float* x, float s, size_t n) {
  for (size_t i = blockIdx.x * 256ull + threadIdx.x; i < n; i += (size_t)gridDim.x * 256ull)
    o[i] = s * x[i];
}
__global__ void ew_bcast(float* o, const float* w, int C, size_t n) {
  for (size_t i = blockIdx.x * 256ull + threadIdx.x; i < n; i += (size_t)gridDim.x * 256ull)
    o[i] = w[i % C];
}
// linear interp 512 -> Ls (align_corners=False), 3 channels
__global__ void ew_interp(float* xs, const float* x, int Ls) {
  size_t total = 512ull * 3 * Ls;
  float scale = 512.f / (float)Ls;
  for (size_t i = blockIdx.x * 256ull + threadIdx.x; i < total; i += (size_t)gridDim.x * 256ull) {
    int j = (int)(i % Ls); size_t bc = i / Ls;
    float pos = (j + 0.5f) * scale - 0.5f;
    pos = fminf(fmaxf(pos, 0.f), 511.f);
    int i0 = (int)floorf(pos); int i1 = (i0 + 1 < 512) ? i0 + 1 : 511;
    float w = pos - (float)i0;
    const float* row = x + bc * 512;
    xs[i] = row[i0] * (1.f - w) + row[i1] * w;
  }
}
// transpose of ew_interp, gather form (deterministic): gx += interpT(g)
__global__ void ew_interpT(float* gx, const float* g, int Ls) {
  size_t total = 512ull * 3 * 512;
  float scale = 512.f / (float)Ls;
  for (size_t i = blockIdx.x * 256ull + threadIdx.x; i < total; i += (size_t)gridDim.x * 256ull) {
    int l = (int)(i % 512); size_t bc = i / 512;
    const float* row = g + bc * Ls;
    float jc = ((float)l + 0.5f) / scale - 0.5f;
    int j0 = (int)floorf(jc) - 2; if (j0 < 0) j0 = 0;
    int j1 = (int)floorf(jc) + 3; if (j1 > Ls - 1) j1 = Ls - 1;
    float a = 0.f;
    for (int j = j0; j <= j1; ++j) {
      float pos = (j + 0.5f) * scale - 0.5f;
      pos = fminf(fmaxf(pos, 0.f), 511.f);
      int i0 = (int)floorf(pos); int i1 = (i0 + 1 < 512) ? i0 + 1 : 511;
      float w = pos - (float)i0;
      if (i0 == l) a += (1.f - w) * row[j];
      if (i1 == l) a += w * row[j];
    }
    gx[i] += a;
  }
}

// ---------------------------------------------------------------------------
extern "C" void kernel_launch(void* const* d_in, const int* in_sizes, int n_in,
                              void* d_out, int out_size, void* d_ws, size_t ws_size,
                              hipStream_t stream) {
  (void)in_sizes; (void)n_in; (void)out_size; (void)ws_size;
  const int B = 512;
  const float* x     = (const float*)d_in[0];
  const float* sigma = (const float*)d_in[1];
  auto P = [&](int i) { return (const float*)d_in[i]; };
  // param indices (setup_inputs insertion order)
  auto tilde_w = [](int i){ return i == 0 ? 2 : 4 + 2*(i-1); };
  auto tilde_b = [](int i){ return i == 0 ? 3 : 5 + 2*(i-1); };
  auto zu_w = [](int i){ return 16 + 2*i; };   auto zu_b = [](int i){ return 17 + 2*i; };
  auto z_w  = [](int i){ return i < 5 ? 28 + i : 33; };
  auto yu_w = [](int i){ return 34 + 2*i; };   auto yu_b = [](int i){ return 35 + 2*i; };
  auto y_w  = [](int i){ return i < 6 ? 48 + i : 54; };
  auto u_w  = [](int i){ return i < 6 ? 55 + 2*i : 67; };
  auto u_b  = [](int i){ return i < 6 ? 56 + 2*i : 68; };
  const int final_w = 69;

  // bump allocator over workspace
  size_t off = 0;
  auto alloc = [&](size_t bytes) -> void* {
    off = (off + 255) & ~(size_t)255;
    void* p = (char*)d_ws + off; off += bytes; return p;
  };
  auto allocF = [&](size_t n) { return (float*)alloc(n * sizeof(float)); };
  auto packW = [&](int pidx, int Co, int Ci, int Kw, bool tr) -> _Float16* {
    size_t n = (size_t)Co * Ci * Kw;
    _Float16* dst = (_Float16*)alloc(n * sizeof(_Float16));
    int blk = (int)((n + 255) / 256); if (blk > 4096) blk = 4096;
    pack_w_kernel<<<blk, 256, 0, stream>>>(P(pidx), dst, Co, Ci, Kw, tr ? 1 : 0);
    return dst;
  };
  auto conv = [&](const float* in, const _Float16* w, const float* bias, float* out,
                  int Cin, int Lin, int Cout, int Lout, int Kw, int stride, int pad, int flags) {
    dim3 g((Lout + 31) / 32, B), blk(256);
    if (Kw == 5) {
      if (flags & F_TRANS) conv1d_wmma<5,  true ><<<g, blk, 0, stream>>>(in, w, bias, out, Cin, Lin, Cout, Lout, stride, pad, flags);
      else                 conv1d_wmma<5,  false><<<g, blk, 0, stream>>>(in, w, bias, out, Cin, Lin, Cout, Lout, stride, pad, flags);
    } else {
      if (flags & F_TRANS) conv1d_wmma<64, true ><<<g, blk, 0, stream>>>(in, w, bias, out, Cin, Lin, Cout, Lout, stride, pad, flags);
      else                 conv1d_wmma<64, false><<<g, blk, 0, stream>>>(in, w, bias, out, Cin, Lin, Cout, Lout, stride, pad, flags);
    }
  };
  auto ewg = [&](size_t n) { size_t b = (n + 255) / 256; return dim3((unsigned)(b > 32768 ? 32768 : b)); };

  const int Lz[7]  = {512, 256, 256, 128, 128, 64, 1};
  const int Lu[6]  = {512, 256, 256, 128, 128, 64};
  const int SZ[6]  = {512, 256, 256, 128, 128, 64};
  const int tstr[6] = {1, 2, 1, 2, 1, 2};
  const int zstr[5] = {2, 1, 2, 1, 2};

  // ---- f16 weight packs (forward + transposed for backward) ----
  _Float16 *wT[7], *wZU[6], *wZ[6], *wZt[6], *wYU[7], *wY[7], *wYt[7], *wU[7];
  for (int i = 1; i <= 6; ++i) wT[i] = packW(tilde_w(i), 64, 64, 5, false);
  for (int i = 0; i < 6; ++i)  wZU[i] = packW(zu_w(i), 128, 64, 5, false);
  for (int i = 0; i < 5; ++i) { wZ[i] = packW(z_w(i), 128, 128, 5, false);
                                wZt[i] = packW(z_w(i), 128, 128, 5, true); }
  wZ[5] = packW(z_w(5), 64, 128, 64, false);  wZt[5] = packW(z_w(5), 64, 128, 64, true);
  for (int i = 0; i < 7; ++i)  wYU[i] = packW(yu_w(i), 3, 64, 5, false);
  for (int i = 0; i < 6; ++i) { wY[i] = packW(y_w(i), 128, 3, 5, false);
                                wYt[i] = packW(y_w(i), 128, 3, 5, true); }
  wY[6] = packW(y_w(6), 64, 3, 64, false);    wYt[6] = packW(y_w(6), 64, 3, 64, true);
  for (int i = 0; i < 6; ++i)  wU[i] = packW(u_w(i), 128, 64, 5, false);
  wU[6] = packW(u_w(6), 64, 64, 64, false);

  // ---- activation buffers ----
  const size_t NU = (size_t)B * 64 * 512, NZ = (size_t)B * 128 * 512, NX = (size_t)B * 3 * 512;
  float *uA = allocF(NU), *uB = allocF(NU);
  float *zA = allocF(NZ), *zB = allocF(NZ), *gt = allocF(NZ);
  float *zpre[7], *rr[6], *yus[7];
  zpre[0] = allocF(NZ);
  for (int i = 1; i <= 5; ++i) zpre[i] = allocF((size_t)B * 128 * Lz[i]);
  zpre[6] = allocF((size_t)B * 64);
  for (int i = 0; i < 6; ++i) rr[i] = allocF((size_t)B * 128 * Lz[i]);
  yus[0] = allocF(NX);
  for (int i = 0; i < 6; ++i) yus[i + 1] = allocF((size_t)B * 3 * Lu[i]);
  float *xs = allocF(NX), *yx = allocF(NX);
  float *dzA = allocF(NZ), *dzB = allocF(NZ);

  // ========================= FORWARD =========================
  ew_tilde0<<<ewg(NU), 256, 0, stream>>>(uA, sigma, P(tilde_w(0)), P(tilde_b(0)), 64ull * 512);
  conv(uA, wYU[0], P(yu_b(0)), yus[0], 64, 512, 3, 512, 5, 1, 2, F_BIAS);
  ew_mul<<<ewg(NX), 256, 0, stream>>>(yx, yus[0], x, NX);
  conv(yx, wY[0], nullptr, zpre[0], 3, 512, 128, 512, 5, 1, 2, 0);
  conv(uA, wU[0], P(u_b(0)), zpre[0], 64, 512, 128, 512, 5, 1, 2, F_BIAS | F_ACC);
  ew_softplus<<<ewg(NZ), 256, 0, stream>>>(zA, zpre[0], NZ);

  float *ucur = uA, *unxt = uB, *zcur = zA, *zscr = zB;
  for (int i = 0; i < 6; ++i) {
    int Lui = (i == 0) ? 512 : Lu[i - 1], Luo = Lu[i];
    int Co = (i < 5) ? 128 : 64;
    int Kz = (i < 5) ? 5 : 64, pz = (i < 5) ? 2 : 0, sz = (i < 5) ? zstr[i] : 1;
    conv(ucur, wT[i + 1], P(tilde_b(i + 1)), unxt, 64, Lui, 64, Luo, 5, tstr[i], 2, F_BIAS | F_SOFTPLUS);
    { float* t = ucur; ucur = unxt; unxt = t; }
    conv(ucur, wZU[i], P(zu_b(i)), rr[i], 64, Luo, 128, Lz[i], 5, 1, 2, F_BIAS | F_RELU);
    ew_mul<<<ewg((size_t)B * 128 * Lz[i]), 256, 0, stream>>>(gt, rr[i], zcur, (size_t)B * 128 * Lz[i]);
    conv(gt, wZ[i], nullptr, zpre[i + 1], 128, Lz[i], Co, Lz[i + 1], Kz, sz, pz, 0);
    ew_interp<<<ewg((size_t)B * 3 * SZ[i]), 256, 0, stream>>>(xs, x, SZ[i]);
    conv(ucur, wYU[i + 1], P(yu_b(i + 1)), yus[i + 1], 64, Luo, 3, SZ[i], 5, 1, 2, F_BIAS);
    ew_mul<<<ewg((size_t)B * 3 * SZ[i]), 256, 0, stream>>>(yx, yus[i + 1], xs, (size_t)B * 3 * SZ[i]);
    conv(yx, wY[i + 1], nullptr, zpre[i + 1], 3, SZ[i], Co, Lz[i + 1], Kz, sz, pz, F_ACC);
    conv(ucur, wU[i + 1], P(u_b(i + 1)), zpre[i + 1], 64, Luo, Co, Lz[i + 1], Kz, sz, pz, F_ACC | F_BIAS);
    if (i < 5) {
      ew_softplus<<<ewg((size_t)B * 128 * Lz[i + 1]), 256, 0, stream>>>(zscr, zpre[i + 1],
                                                                        (size_t)B * 128 * Lz[i + 1]);
      float* t = zcur; zcur = zscr; zscr = t;
    }
  }

  // ========================= BACKWARD (exact: u is x-independent) ==========
  float* gx = (float*)d_out;
  ew_scale<<<ewg(NX), 256, 0, stream>>>(gx, x, 0.2f, NX);  // 2*ALPHA*x
  float *dcur = dzA, *dnxt = dzB;
  ew_bcast<<<ewg((size_t)B * 64), 256, 0, stream>>>(dcur, P(final_w), 64, (size_t)B * 64);
  for (int i = 5; i >= 0; --i) {
    int Co = (i < 5) ? 128 : 64;
    int Kz = (i < 5) ? 5 : 64, pz = (i < 5) ? 2 : 0, sz = (i < 5) ? zstr[i] : 1;
    size_t nz = (size_t)B * Co * Lz[i + 1];
    ew_sigmul<<<ewg(nz), 256, 0, stream>>>(gt, dcur, zpre[i + 1], nz);       // dz_pre
    conv(gt, wZt[i], nullptr, dnxt, Co, Lz[i + 1], 128, Lz[i], Kz, sz, pz, F_TRANS);   // dg
    conv(gt, wYt[i + 1], nullptr, yx, Co, Lz[i + 1], 3, SZ[i], Kz, sz, pz, F_TRANS);   // d(yu*xs)
    ew_mul<<<ewg((size_t)B * 3 * SZ[i]), 256, 0, stream>>>(xs, yx, yus[i + 1], (size_t)B * 3 * SZ[i]);
    ew_interpT<<<ewg(NX), 256, 0, stream>>>(gx, xs, SZ[i]);                  // gx += interpT(dxs)
    ew_mul<<<ewg((size_t)B * 128 * Lz[i]), 256, 0, stream>>>(dcur, dnxt, rr[i], (size_t)B * 128 * Lz[i]);
  }
  // initial block: gx += yu0 * convT_y0(dz0_pre)
  ew_sigmul<<<ewg(NZ), 256, 0, stream>>>(gt, dcur, zpre[0], NZ);
  conv(gt, wYt[0], nullptr, yx, 128, 512, 3, 512, 5, 1, 2, F_TRANS);
  ew_mul_acc<<<ewg(NX), 256, 0, stream>>>(gx, yx, yus[0], NX);
}
